// ResidualRecursiveCell_24902220382489
// MI455X (gfx1250) — matrix operand order repaired and checked by reading, not compiled, not run on hardware
//
#include <hip/hip_runtime.h>
#include <hip/hip_bf16.h>

#define NE 4
#define EPSV 1e-6f
#define RESW 0.1f
#define ATTN_SCALE 0.125f   // 64^-0.5
#define W1 8
#define W2 8

typedef __bf16 bf16_t;
typedef __attribute__((ext_vector_type(16))) __bf16 v16bf;
typedef __attribute__((ext_vector_type(8)))  __bf16 v8bf;
typedef __attribute__((ext_vector_type(8)))  float  v8f;

__device__ __forceinline__ v8f zero8() {
    v8f v;
#pragma unroll
    for (int i = 0; i < 8; i++) v[i] = 0.0f;
    return v;
}

__device__ __forceinline__ v8f wmma_bf(v16bf a, v16bf b, v8f c) {
    // D = A(16x32 bf16) * B(32x16 bf16) + C(16x16 f32)
    return __builtin_amdgcn_wmma_f32_16x16x32_bf16(false, a, false, b, (short)0, c, false, false);
}

// A fragment (16x32) from f32 row-major source, converting to bf16.
// lane<16: K = kb+[0..7] and kb+[16..23]; lane>=16: +8 on both halves.
__device__ __forceinline__ v16bf a_from_f32(const float* base, int ld, int kb, int lane) {
    int m  = lane & 15;
    int kh = (lane >> 4) << 3;
    const float* p = base + m * ld + kb + kh;
    v16bf a;
#pragma unroll
    for (int i = 0; i < 8; i++) { a[i] = (bf16_t)p[i]; a[i + 8] = (bf16_t)p[i + 16]; }
    return a;
}

// A fragment (16x32) from bf16 row-major source (global or LDS).
__device__ __forceinline__ v16bf a_frag(const bf16_t* base, int ld, int kb, int lane) {
    int m  = lane & 15;
    int kh = (lane >> 4) << 3;
    const bf16_t* p = base + m * ld + kb + kh;
    v8bf lo = *(const v8bf*)p;
    v8bf hi = *(const v8bf*)(p + 16);
    return __builtin_shufflevector(lo, hi, 0,1,2,3,4,5,6,7,8,9,10,11,12,13,14,15);
}

// B fragment (32x16) for y = x @ W.T with W row-major [out][in]:
// B(k,n) = W[nbase+n][kb+k]; lane<16: K = kb+[0..7] and kb+[16..23]; lane>=16: +8 on both halves.
__device__ __forceinline__ v16bf b_frag(const bf16_t* W, int ld, int nbase, int kb, int lane) {
    int n = lane & 15;
    int k = kb + ((lane >> 4) << 3);
    const bf16_t* p = W + (size_t)(nbase + n) * ld + k;
    v8bf lo = *(const v8bf*)p;
    v8bf hi = *(const v8bf*)(p + 16);
    return __builtin_shufflevector(lo, hi, 0,1,2,3,4,5,6,7,8,9,10,11,12,13,14,15);
}

// reduce across the 16-lane half-groups (xor 1,2,4,8 never crosses bit 4)
__device__ __forceinline__ float redsum16(float v) {
#pragma unroll
    for (int o = 1; o < 16; o <<= 1) v += __shfl_xor(v, o, 32);
    return v;
}
__device__ __forceinline__ float redmax16(float v) {
#pragma unroll
    for (int o = 1; o < 16; o <<= 1) v = fmaxf(v, __shfl_xor(v, o, 32));
    return v;
}

__global__ void cvt_kernel(const float* __restrict__ src, bf16_t* __restrict__ dst,
                           int n, float scale) {
    int i = blockIdx.x * blockDim.x + threadIdx.x;
    if (i < n) dst[i] = (bf16_t)(src[i] * scale);
}

// ---------------- Stage 1: gate -> MoE -> LN -> ModReLU -> QKV/ctx/res ----------------
__global__ __launch_bounds__(32 * W1)
void stage1_kernel(const float* __restrict__ zr, const float* __restrict__ zi,
                   const float* __restrict__ gateW, const float* __restrict__ gateB,
                   const float* __restrict__ lnScale, const float* __restrict__ lnShift,
                   const float* __restrict__ modBias, const float* __restrict__ ctxB,
                   const float* __restrict__ resB,
                   const bf16_t* __restrict__ eWr, const bf16_t* __restrict__ eWi,
                   const bf16_t* __restrict__ eWiN,
                   const bf16_t* __restrict__ qWr, const bf16_t* __restrict__ qWi,
                   const bf16_t* __restrict__ qWiN,
                   const bf16_t* __restrict__ kWr, const bf16_t* __restrict__ kWi,
                   const bf16_t* __restrict__ kWiN,
                   const bf16_t* __restrict__ vWr, const bf16_t* __restrict__ vWi,
                   const bf16_t* __restrict__ vWiN,
                   const bf16_t* __restrict__ ctxW, const bf16_t* __restrict__ resW,
                   bf16_t* __restrict__ qfO, bf16_t* __restrict__ kfO,
                   bf16_t* __restrict__ vrT, bf16_t* __restrict__ viT,
                   float* __restrict__ resO, int N) {
    __shared__ bf16_t sAr[W1][16 * 64];
    __shared__ bf16_t sAi[W1][16 * 64];
    __shared__ float  sGw[W1][16 * NE];

    const int lane = threadIdx.x & 31;
    const int wid  = threadIdx.x >> 5;
    const int row0 = (blockIdx.x * W1 + wid) * 16;
    const int rb   = (lane & 16) ? 8 : 0;
    const int col  = lane & 15;

    // ---- gate softmax (lanes 0..15, one row each) ----
    if (lane < 16) {
        int row = row0 + lane;
        float lg[NE];
#pragma unroll
        for (int e = 0; e < NE; e++) {
            float s = gateB[e];
            const float* gw = gateW + e * 128;
            for (int d = 0; d < 64; d++)
                s += zr[(size_t)row * 64 + d] * gw[d] + zi[(size_t)row * 64 + d] * gw[64 + d];
            lg[e] = s;
        }
        float mx = fmaxf(fmaxf(lg[0], lg[1]), fmaxf(lg[2], lg[3]));
        float ex[NE], den = 0.0f;
#pragma unroll
        for (int e = 0; e < NE; e++) { ex[e] = __expf(lg[e] - mx); den += ex[e]; }
#pragma unroll
        for (int e = 0; e < NE; e++) sGw[wid][lane * NE + e] = ex[e] / den;
    }
    asm volatile("s_wait_dscnt 0" ::: "memory");

    // ---- A fragments of z (bf16) ----
    const float* zrt = zr + (size_t)row0 * 64;
    const float* zit = zi + (size_t)row0 * 64;
    v16bf azr0 = a_from_f32(zrt, 64, 0, lane);
    v16bf azr1 = a_from_f32(zrt, 64, 32, lane);
    v16bf azi0 = a_from_f32(zit, 64, 0, lane);
    v16bf azi1 = a_from_f32(zit, 64, 32, lane);

    // ---- MoE: mr/mi = sum_e gw[e] * (complex expert linear) ----
    v8f mr[4], mi[4];
#pragma unroll
    for (int nt = 0; nt < 4; nt++) { mr[nt] = zero8(); mi[nt] = zero8(); }
#pragma unroll
    for (int e = 0; e < NE; e++) {
        const bf16_t* Wr  = eWr  + e * 4096;
        const bf16_t* Wi  = eWi  + e * 4096;
        const bf16_t* WiN = eWiN + e * 4096;
        float g[8];
#pragma unroll
        for (int r = 0; r < 8; r++) g[r] = sGw[wid][(rb + r) * NE + e];
#pragma unroll
        for (int nt = 0; nt < 4; nt++) {
            v8f er = zero8();
            er = wmma_bf(azr0, b_frag(Wr, 64, nt * 16, 0, lane), er);
            er = wmma_bf(azr1, b_frag(Wr, 64, nt * 16, 32, lane), er);
            er = wmma_bf(azi0, b_frag(WiN, 64, nt * 16, 0, lane), er);
            er = wmma_bf(azi1, b_frag(WiN, 64, nt * 16, 32, lane), er);
            v8f ei = zero8();
            ei = wmma_bf(azi0, b_frag(Wr, 64, nt * 16, 0, lane), ei);
            ei = wmma_bf(azi1, b_frag(Wr, 64, nt * 16, 32, lane), ei);
            ei = wmma_bf(azr0, b_frag(Wi, 64, nt * 16, 0, lane), ei);
            ei = wmma_bf(azr1, b_frag(Wi, 64, nt * 16, 32, lane), ei);
#pragma unroll
            for (int r = 0; r < 8; r++) { mr[nt][r] += g[r] * er[r]; mi[nt][r] += g[r] * ei[r]; }
        }
    }

    // ---- ComplexLayerNorm (magnitude stats, ddof=1) + ModReLU ----
    float magv[4][8], mean[8], inv[8];
    float su[8];
#pragma unroll
    for (int r = 0; r < 8; r++) su[r] = 0.0f;
#pragma unroll
    for (int nt = 0; nt < 4; nt++)
#pragma unroll
        for (int r = 0; r < 8; r++) {
            float m = sqrtf(mr[nt][r] * mr[nt][r] + mi[nt][r] * mi[nt][r]) + EPSV;
            magv[nt][r] = m;
            su[r] += m;
        }
#pragma unroll
    for (int r = 0; r < 8; r++) mean[r] = redsum16(su[r]) * (1.0f / 64.0f);
#pragma unroll
    for (int r = 0; r < 8; r++) su[r] = 0.0f;
#pragma unroll
    for (int nt = 0; nt < 4; nt++)
#pragma unroll
        for (int r = 0; r < 8; r++) {
            float d = magv[nt][r] - mean[r];
            su[r] += d * d;
        }
#pragma unroll
    for (int r = 0; r < 8; r++) {
        float var = redsum16(su[r]) * (1.0f / 63.0f);
        inv[r] = 1.0f / sqrtf(var + EPSV);
    }
#pragma unroll
    for (int nt = 0; nt < 4; nt++) {
        float scl = lnScale[nt * 16 + col];
        float sft = lnShift[nt * 16 + col];
        float mb  = modBias[nt * 16 + col];
#pragma unroll
        for (int r = 0; r < 8; r++) {
            float nm = (magv[nt][r] - mean[r]) * inv[r] * scl + sft;
            float hr = mr[nt][r], hi = mi[nt][r];
            float rad = sqrtf(hr * hr + hi * hi);
            float ir = (rad > 0.0f) ? (1.0f / rad) : 0.0f;
            float cp = (rad > 0.0f) ? hr * ir : 1.0f;
            float sp = hi * ir;
            float lr_ = nm * cp, li_ = nm * sp;
            float nrm = sqrtf(lr_ * lr_ + li_ * li_) + EPSV;
            float s = fmaxf(nrm + mb, 0.0f) / nrm;
            sAr[wid][(rb + r) * 64 + nt * 16 + col] = (bf16_t)(lr_ * s);
            sAi[wid][(rb + r) * 64 + nt * 16 + col] = (bf16_t)(li_ * s);
        }
    }
    asm volatile("s_wait_dscnt 0" ::: "memory");

    v16bf aar0 = a_frag(sAr[wid], 64, 0, lane);
    v16bf aar1 = a_frag(sAr[wid], 64, 32, lane);
    v16bf aai0 = a_frag(sAi[wid], 64, 0, lane);
    v16bf aai1 = a_frag(sAi[wid], 64, 32, lane);

    // ---- K projection (store immediately) ----
#pragma unroll
    for (int nt = 0; nt < 4; nt++) {
        v8f rr = zero8();
        rr = wmma_bf(aar0, b_frag(kWr, 64, nt * 16, 0, lane), rr);
        rr = wmma_bf(aar1, b_frag(kWr, 64, nt * 16, 32, lane), rr);
        rr = wmma_bf(aai0, b_frag(kWiN, 64, nt * 16, 0, lane), rr);
        rr = wmma_bf(aai1, b_frag(kWiN, 64, nt * 16, 32, lane), rr);
        v8f ri = zero8();
        ri = wmma_bf(aai0, b_frag(kWr, 64, nt * 16, 0, lane), ri);
        ri = wmma_bf(aai1, b_frag(kWr, 64, nt * 16, 32, lane), ri);
        ri = wmma_bf(aar0, b_frag(kWi, 64, nt * 16, 0, lane), ri);
        ri = wmma_bf(aar1, b_frag(kWi, 64, nt * 16, 32, lane), ri);
#pragma unroll
        for (int r = 0; r < 8; r++) {
            int row = row0 + rb + r;
            kfO[(size_t)row * 128 + nt * 16 + col] = (bf16_t)rr[r];
            kfO[(size_t)row * 128 + 64 + nt * 16 + col] = (bf16_t)ri[r];
        }
    }
    // ---- V projection (store transposed: [feat][N]) ----
#pragma unroll
    for (int nt = 0; nt < 4; nt++) {
        v8f rr = zero8();
        rr = wmma_bf(aar0, b_frag(vWr, 64, nt * 16, 0, lane), rr);
        rr = wmma_bf(aar1, b_frag(vWr, 64, nt * 16, 32, lane), rr);
        rr = wmma_bf(aai0, b_frag(vWiN, 64, nt * 16, 0, lane), rr);
        rr = wmma_bf(aai1, b_frag(vWiN, 64, nt * 16, 32, lane), rr);
        v8f ri = zero8();
        ri = wmma_bf(aai0, b_frag(vWr, 64, nt * 16, 0, lane), ri);
        ri = wmma_bf(aai1, b_frag(vWr, 64, nt * 16, 32, lane), ri);
        ri = wmma_bf(aar0, b_frag(vWi, 64, nt * 16, 0, lane), ri);
        ri = wmma_bf(aar1, b_frag(vWi, 64, nt * 16, 32, lane), ri);
#pragma unroll
        for (int r = 0; r < 8; r++) {
            int row = row0 + rb + r;
            vrT[(size_t)(nt * 16 + col) * N + row] = (bf16_t)rr[r];
            viT[(size_t)(nt * 16 + col) * N + row] = (bf16_t)ri[r];
        }
    }
    // ---- Q projection (kept in registers for ctx gating) ----
    v8f qr_[4], qi_[4];
#pragma unroll
    for (int nt = 0; nt < 4; nt++) {
        v8f rr = zero8();
        rr = wmma_bf(aar0, b_frag(qWr, 64, nt * 16, 0, lane), rr);
        rr = wmma_bf(aar1, b_frag(qWr, 64, nt * 16, 32, lane), rr);
        rr = wmma_bf(aai0, b_frag(qWiN, 64, nt * 16, 0, lane), rr);
        rr = wmma_bf(aai1, b_frag(qWiN, 64, nt * 16, 32, lane), rr);
        qr_[nt] = rr;
        v8f ri = zero8();
        ri = wmma_bf(aai0, b_frag(qWr, 64, nt * 16, 0, lane), ri);
        ri = wmma_bf(aai1, b_frag(qWr, 64, nt * 16, 32, lane), ri);
        ri = wmma_bf(aar0, b_frag(qWi, 64, nt * 16, 0, lane), ri);
        ri = wmma_bf(aar1, b_frag(qWi, 64, nt * 16, 32, lane), ri);
        qi_[nt] = ri;
    }
    // ---- ctx gate (sigmoid(zf@ctxW.T+b)) applied to qf; res = zf@resW.T+b ----
#pragma unroll
    for (int nt = 0; nt < 8; nt++) {
        v8f c = zero8();
        c = wmma_bf(azr0, b_frag(ctxW, 128, nt * 16, 0, lane), c);
        c = wmma_bf(azr1, b_frag(ctxW, 128, nt * 16, 32, lane), c);
        c = wmma_bf(azi0, b_frag(ctxW, 128, nt * 16, 64, lane), c);
        c = wmma_bf(azi1, b_frag(ctxW, 128, nt * 16, 96, lane), c);
        v8f rs = zero8();
        rs = wmma_bf(azr0, b_frag(resW, 128, nt * 16, 0, lane), rs);
        rs = wmma_bf(azr1, b_frag(resW, 128, nt * 16, 32, lane), rs);
        rs = wmma_bf(azi0, b_frag(resW, 128, nt * 16, 64, lane), rs);
        rs = wmma_bf(azi1, b_frag(resW, 128, nt * 16, 96, lane), rs);
        float cb = ctxB[nt * 16 + col];
        float rbv = resB[nt * 16 + col];
#pragma unroll
        for (int r = 0; r < 8; r++) {
            int row = row0 + rb + r;
            float cw = 1.0f / (1.0f + __expf(-(c[r] + cb)));
            float qv = (nt < 4) ? qr_[nt][r] : qi_[nt - 4][r];
            qfO[(size_t)row * 128 + nt * 16 + col] = (bf16_t)(qv * cw);
            resO[(size_t)row * 128 + nt * 16 + col] = rs[r] + rbv;
        }
    }
}

// ---------------- Stage 2: flash attention (64-key tiles) + residual + heads ----------------
__global__ __launch_bounds__(32 * W2)
void attn_kernel(const bf16_t* __restrict__ qf, const bf16_t* __restrict__ kf,
                 const bf16_t* __restrict__ vrT, const bf16_t* __restrict__ viT,
                 const float* __restrict__ resO,
                 const float* __restrict__ haltW, const float* __restrict__ haltB,
                 const float* __restrict__ stackW, const float* __restrict__ stackB,
                 float* __restrict__ out, int N) {
    __shared__ bf16_t sP[W2][16 * 64];

    const int lane = threadIdx.x & 31;
    const int wid  = threadIdx.x >> 5;
    const int row0 = (blockIdx.x * W2 + wid) * 16;
    const int rb   = (lane & 16) ? 8 : 0;
    const int col  = lane & 15;

    v16bf aq0 = a_frag(qf + (size_t)row0 * 128, 128, 0, lane);
    v16bf aq1 = a_frag(qf + (size_t)row0 * 128, 128, 32, lane);
    v16bf aq2 = a_frag(qf + (size_t)row0 * 128, 128, 64, lane);
    v16bf aq3 = a_frag(qf + (size_t)row0 * 128, 128, 96, lane);

    v8f o[8];
#pragma unroll
    for (int f = 0; f < 8; f++) o[f] = zero8();
    float mrow[8], lrow[8];
#pragma unroll
    for (int r = 0; r < 8; r++) { mrow[r] = -3.0e38f; lrow[r] = 0.0f; }

    for (int kb0 = 0; kb0 < N; kb0 += 64) {
        if (kb0 + 64 < N) __builtin_prefetch(kf + (size_t)(kb0 + 64) * 128, 0, 1);
        // S = Q @ K^T, four 16x16 tiles covering 64 keys
        v8f s[4];
#pragma unroll
        for (int j = 0; j < 4; j++) {
            v8f acc = zero8();
            acc = wmma_bf(aq0, b_frag(kf, 128, kb0 + j * 16, 0, lane), acc);
            acc = wmma_bf(aq1, b_frag(kf, 128, kb0 + j * 16, 32, lane), acc);
            acc = wmma_bf(aq2, b_frag(kf, 128, kb0 + j * 16, 64, lane), acc);
            acc = wmma_bf(aq3, b_frag(kf, 128, kb0 + j * 16, 96, lane), acc);
            s[j] = acc;
        }

        float al[8];
#pragma unroll
        for (int r = 0; r < 8; r++) {
            float a0 = s[0][r] * ATTN_SCALE;
            float a1 = s[1][r] * ATTN_SCALE;
            float a2 = s[2][r] * ATTN_SCALE;
            float a3 = s[3][r] * ATTN_SCALE;
            float tm = redmax16(fmaxf(fmaxf(a0, a1), fmaxf(a2, a3)));
            float mn = fmaxf(mrow[r], tm);
            al[r] = __expf(mrow[r] - mn);
            mrow[r] = mn;
            float p0 = __expf(a0 - mn);
            float p1 = __expf(a1 - mn);
            float p2 = __expf(a2 - mn);
            float p3 = __expf(a3 - mn);
            sP[wid][(rb + r) * 64 + col]      = (bf16_t)p0;
            sP[wid][(rb + r) * 64 + 16 + col] = (bf16_t)p1;
            sP[wid][(rb + r) * 64 + 32 + col] = (bf16_t)p2;
            sP[wid][(rb + r) * 64 + 48 + col] = (bf16_t)p3;
            lrow[r] = lrow[r] * al[r] + redsum16((p0 + p1) + (p2 + p3));
        }
#pragma unroll
        for (int f = 0; f < 8; f++)
#pragma unroll
            for (int r = 0; r < 8; r++) o[f][r] *= al[r];

        asm volatile("s_wait_dscnt 0" ::: "memory");
        v16bf aP0 = a_frag(sP[wid], 64, 0, lane);
        v16bf aP1 = a_frag(sP[wid], 64, 32, lane);
#pragma unroll
        for (int f = 0; f < 4; f++) {
            o[f]     = wmma_bf(aP0, b_frag(vrT + kb0,      N, f * 16, 0, lane), o[f]);
            o[f]     = wmma_bf(aP1, b_frag(vrT + kb0 + 32, N, f * 16, 0, lane), o[f]);
            o[4 + f] = wmma_bf(aP0, b_frag(viT + kb0,      N, f * 16, 0, lane), o[4 + f]);
            o[4 + f] = wmma_bf(aP1, b_frag(viT + kb0 + 32, N, f * 16, 0, lane), o[4 + f]);
        }
    }

    // epilogue: 1/l, residual, norm, heads
    float invl[8];
#pragma unroll
    for (int r = 0; r < 8; r++) invl[r] = 1.0f / lrow[r];

    float pr[4][8], pi[4][8], ss[8];
#pragma unroll
    for (int r = 0; r < 8; r++) ss[r] = 0.0f;
#pragma unroll
    for (int f = 0; f < 4; f++)
#pragma unroll
        for (int r = 0; r < 8; r++) {
            int row = row0 + rb + r;
            float a = o[f][r] * invl[r]     + RESW * resO[(size_t)row * 128 + f * 16 + col];
            float b = o[4 + f][r] * invl[r] + RESW * resO[(size_t)row * 128 + 64 + f * 16 + col];
            pr[f][r] = a; pi[f][r] = b;
            ss[r] += a * a + b * b;
        }
#pragma unroll
    for (int r = 0; r < 8; r++) ss[r] = 1.0f / (sqrtf(redsum16(ss[r])) + 1e-6f);

    float* outPr = out;
    float* outPi = out + (size_t)N * 64;
    float* outH  = out + (size_t)N * 128;
    float* outS  = out + (size_t)N * 128 + N;
#pragma unroll
    for (int f = 0; f < 4; f++)
#pragma unroll
        for (int r = 0; r < 8; r++) {
            int row = row0 + rb + r;
            outPr[(size_t)row * 64 + f * 16 + col] = pr[f][r];
            outPi[(size_t)row * 64 + f * 16 + col] = pi[f][r];
        }

    float hwr[4], hwi[4];
#pragma unroll
    for (int f = 0; f < 4; f++) { hwr[f] = haltW[f * 16 + col]; hwi[f] = haltW[64 + f * 16 + col]; }
#pragma unroll
    for (int r = 0; r < 8; r++) {
        float h = 0.0f;
#pragma unroll
        for (int f = 0; f < 4; f++) h += pr[f][r] * ss[r] * hwr[f] + pi[f][r] * ss[r] * hwi[f];
        h = redsum16(h);
        if (col == 0) outH[row0 + rb + r] = h + haltB[0];
    }
#pragma unroll
    for (int j = 0; j < 3; j++) {
        float swr[4], swi[4];
#pragma unroll
        for (int f = 0; f < 4; f++) {
            swr[f] = stackW[j * 128 + f * 16 + col];
            swi[f] = stackW[j * 128 + 64 + f * 16 + col];
        }
#pragma unroll
        for (int r = 0; r < 8; r++) {
            float h = 0.0f;
#pragma unroll
            for (int f = 0; f < 4; f++) h += pr[f][r] * ss[r] * swr[f] + pi[f][r] * ss[r] * swi[f];
            h = redsum16(h);
            if (col == 0) outS[(size_t)(row0 + rb + r) * 3 + j] = h + stackB[j];
        }
    }
}

extern "C" void kernel_launch(void* const* d_in, const int* in_sizes, int n_in,
                              void* d_out, int out_size, void* d_ws, size_t ws_size,
                              hipStream_t stream) {
    const float* zr    = (const float*)d_in[0];
    const float* zi    = (const float*)d_in[1];
    const float* gateW = (const float*)d_in[2];
    const float* gateB = (const float*)d_in[3];
    const float* expWr = (const float*)d_in[4];
    const float* expWi = (const float*)d_in[5];
    const float* lnS   = (const float*)d_in[6];
    const float* lnSh  = (const float*)d_in[7];
    const float* modB  = (const float*)d_in[8];
    const float* qWr   = (const float*)d_in[9];
    const float* qWi   = (const float*)d_in[10];
    const float* kWr   = (const float*)d_in[11];
    const float* kWi   = (const float*)d_in[12];
    const float* vWr   = (const float*)d_in[13];
    const float* vWi   = (const float*)d_in[14];
    const float* ctxW  = (const float*)d_in[15];
    const float* ctxB  = (const float*)d_in[16];
    const float* resW  = (const float*)d_in[17];
    const float* resB  = (const float*)d_in[18];
    const float* haltW = (const float*)d_in[19];
    const float* haltB = (const float*)d_in[20];
    const float* stackW= (const float*)d_in[21];
    const float* stackB= (const float*)d_in[22];
    const int N = in_sizes[0] / 64;

    // workspace layout: bf16 weight pack, then activations
    bf16_t* w = (bf16_t*)d_ws;
    bf16_t* eWr  = w + 0;       // 4*64*64
    bf16_t* eWi  = w + 16384;
    bf16_t* eWiN = w + 32768;
    bf16_t* pQr  = w + 49152;   // 64*64 each
    bf16_t* pQi  = w + 53248;
    bf16_t* pQiN = w + 57344;
    bf16_t* pKr  = w + 61440;
    bf16_t* pKi  = w + 65536;
    bf16_t* pKiN = w + 69632;
    bf16_t* pVr  = w + 73728;
    bf16_t* pVi  = w + 77824;
    bf16_t* pViN = w + 81920;
    bf16_t* pCtx = w + 86016;   // 128*128
    bf16_t* pRes = w + 102400;  // 128*128
    char* actb = (char*)d_ws + 262144;
    bf16_t* qf  = (bf16_t*)actb;                                        // N*128 bf16
    bf16_t* kf  = (bf16_t*)(actb + (size_t)N * 128 * 2);                // N*128 bf16
    bf16_t* vrT = (bf16_t*)(actb + (size_t)N * 128 * 4);                // 64*N bf16
    bf16_t* viT = (bf16_t*)(actb + (size_t)N * 128 * 4 + (size_t)N * 64 * 2);
    float*  rsO = (float*) (actb + (size_t)N * 128 * 4 + (size_t)N * 64 * 4); // N*128 f32

    cvt_kernel<<<64, 256, 0, stream>>>(expWr, eWr, 16384, 1.0f);
    cvt_kernel<<<64, 256, 0, stream>>>(expWi, eWi, 16384, 1.0f);
    cvt_kernel<<<64, 256, 0, stream>>>(expWi, eWiN, 16384, -1.0f);
    cvt_kernel<<<16, 256, 0, stream>>>(qWr, pQr, 4096, 1.0f);
    cvt_kernel<<<16, 256, 0, stream>>>(qWi, pQi, 4096, 1.0f);
    cvt_kernel<<<16, 256, 0, stream>>>(qWi, pQiN, 4096, -1.0f);
    cvt_kernel<<<16, 256, 0, stream>>>(kWr, pKr, 4096, 1.0f);
    cvt_kernel<<<16, 256, 0, stream>>>(kWi, pKi, 4096, 1.0f);
    cvt_kernel<<<16, 256, 0, stream>>>(kWi, pKiN, 4096, -1.0f);
    cvt_kernel<<<16, 256, 0, stream>>>(vWr, pVr, 4096, 1.0f);
    cvt_kernel<<<16, 256, 0, stream>>>(vWi, pVi, 4096, 1.0f);
    cvt_kernel<<<16, 256, 0, stream>>>(vWi, pViN, 4096, -1.0f);
    cvt_kernel<<<64, 256, 0, stream>>>(ctxW, pCtx, 16384, 1.0f);
    cvt_kernel<<<64, 256, 0, stream>>>(resW, pRes, 16384, 1.0f);

    const int blocks = N / (16 * W1);   // 64 for N=8192
    stage1_kernel<<<blocks, 32 * W1, 0, stream>>>(
        zr, zi, gateW, gateB, lnS, lnSh, modB, ctxB, resB,
        eWr, eWi, eWiN, pQr, pQi, pQiN, pKr, pKi, pKiN, pVr, pVi, pViN,
        pCtx, pRes, qf, kf, vrT, viT, rsO, N);

    attn_kernel<<<N / (16 * W2), 32 * W2, 0, stream>>>(
        qf, kf, vrT, viT, rsO, haltW, haltB, stackW, stackB, (float*)d_out, N);
}
